// BigramLanguageModel_27410481283474
// MI455X (gfx1250) — compile-verified, matrix operand order checked by reading, not run on hardware
//
#include <hip/hip_runtime.h>
#include <hip/hip_bf16.h>
#include <stdint.h>

#define DEVINL __device__ __forceinline__

typedef __attribute__((ext_vector_type(8)))  __bf16 bf16x8;
typedef __attribute__((ext_vector_type(16))) __bf16 bf16x16;
typedef __attribute__((ext_vector_type(8)))  float  floatx8;
typedef __attribute__((ext_vector_type(4)))  int    i32x4;

constexpr int kV = 50257, kC = 384, kT = 1024, kH = 6, kHS = 64, kNL = 3, kB = 2;
constexpr int kM = kB * kT;  // 2048 activation rows
constexpr int kLdsStride = 40;  // elements per row in LDS A-tile (64B data + 16B pad)

// ---------------------------------------------------------------- helpers
DEVINL __bf16 f2bf(float f) {
  unsigned u = __builtin_bit_cast(unsigned, f);
  unsigned r = u + 0x7FFFu + ((u >> 16) & 1u);  // round-to-nearest-even
  unsigned short h = (unsigned short)(r >> 16);
  return __builtin_bit_cast(__bf16, h);
}

// 16-bit A/B fragment per ISA 7.12.2: lane L holds row (L&15); K pairs are
// kb..kb+7 and kb+16..kb+23 where kb = k0 + 8*(L>=16). Two 16B loads.
DEVINL bf16x16 load_frag(const __bf16* p) {
  bf16x8 lo = *(const bf16x8*)(p);
  bf16x8 hi = *(const bf16x8*)(p + 16);
  return __builtin_shufflevector(lo, hi, 0,1,2,3,4,5,6,7,8,9,10,11,12,13,14,15);
}

// Async global -> LDS copy of 16B (CDNA5 ASYNCcnt path).
// Builtin signature (from hipcc diagnostic): param0 is AS(1) i32x4*.
DEVINL void async_cp16(const __bf16* gsrc, __bf16* ldst) {
#if __has_builtin(__builtin_amdgcn_global_load_async_to_lds_b128)
  __builtin_amdgcn_global_load_async_to_lds_b128(
      (__attribute__((address_space(1))) i32x4*)gsrc,
      (__attribute__((address_space(3))) i32x4*)ldst, 0, 0);
#else
  unsigned loff = (unsigned)(uintptr_t)ldst;  // low 32 bits == LDS offset
  asm volatile("global_load_async_to_lds_b128 %0, %1, off"
               :: "v"(loff), "v"(gsrc) : "memory");
#endif
}

DEVINL void wait_async0() {
#if __has_builtin(__builtin_amdgcn_s_wait_asynccnt)
  __builtin_amdgcn_s_wait_asynccnt(0);
#else
  asm volatile("s_wait_asynccnt 0x0" ::: "memory");
#endif
}

// ---------------------------------------------------------------- embed
__global__ void k_embed(const int* __restrict__ x, const float* __restrict__ tok,
                        const float* __restrict__ pos, float* __restrict__ h) {
  long i = (long)blockIdx.x * blockDim.x + threadIdx.x;
  if (i >= (long)kM * kC) return;
  int c = (int)(i % kC);
  long bt = i / kC;
  int t = (int)(bt % kT);
  h[i] = tok[(long)x[bt] * kC + c] + pos[(long)t * kC + c];
}

// ---------------------------------------------------------------- layernorm -> bf16
__global__ void k_layernorm(const float* __restrict__ h, const float* __restrict__ g,
                            const float* __restrict__ b, __bf16* __restrict__ out) {
  __shared__ float ssum[128], ssq[128];
  int row = blockIdx.x, tid = threadIdx.x;
  const float* p = h + (long)row * kC;
  float s = 0.f, q = 0.f;
  for (int c = tid; c < kC; c += 128) { float v = p[c]; s += v; q += v * v; }
  ssum[tid] = s; ssq[tid] = q;
  __syncthreads();
  for (int st = 64; st > 0; st >>= 1) {
    if (tid < st) { ssum[tid] += ssum[tid + st]; ssq[tid] += ssq[tid + st]; }
    __syncthreads();
  }
  float mu  = ssum[0] / kC;
  float var = ssq[0] / kC - mu * mu;
  float r   = rsqrtf(var + 1e-5f);
  for (int c = tid; c < kC; c += 128)
    out[(long)row * kC + c] = f2bf((p[c] - mu) * r * g[c] + b[c]);
}

// ---------------------------------------------------------------- weight converts
// Generic: W (K x N, f32, row-major) -> Wt (N x K, bf16)
__global__ void k_transpose_w(const float* __restrict__ W, __bf16* __restrict__ Wt,
                              int K, int N) {
  long i = (long)blockIdx.x * blockDim.x + threadIdx.x;
  if (i >= (long)K * N) return;
  int n = (int)(i % N);
  int k = (int)(i / N);
  Wt[(long)n * K + k] = f2bf(W[i]);
}

// Wq/Wk/Wv per layer: (H, C, HS) -> Wt[(h*HS+s)*C + c]
__global__ void k_conv_qkv(const float* __restrict__ W, __bf16* __restrict__ Wt) {
  long i = (long)blockIdx.x * blockDim.x + threadIdx.x;
  if (i >= (long)kH * kC * kHS) return;
  int s = (int)(i % kHS);
  int c = (int)((i / kHS) % kC);
  int h = (int)(i / ((long)kHS * kC));
  Wt[((long)h * kHS + s) * kC + c] = f2bf(W[i]);
}

__global__ void k_f32_to_bf16(const float* __restrict__ in, __bf16* __restrict__ out, long n) {
  long i = (long)blockIdx.x * blockDim.x + threadIdx.x;
  if (i < n) out[i] = f2bf(in[i]);
}

// ---------------------------------------------------------------- softmax over QUERY axis
// Reference does softmax(w, axis=-2) after causal mask: per column tk, over rows tq>=tk.
__global__ void k_softmax_q(const float* __restrict__ w, __bf16* __restrict__ o, float scale) {
  int idx = blockIdx.x * blockDim.x + threadIdx.x;  // over B*H*T columns
  if (idx >= kB * kH * kT) return;
  int tk = idx % kT;
  long bh = idx / kT;
  const float* p = w + bh * (long)kT * kT + tk;
  __bf16* po = o + bh * (long)kT * kT + tk;
  float mx = -3.4e38f;
  for (int tq = tk; tq < kT; ++tq) mx = fmaxf(mx, p[(long)tq * kT] * scale);
  float s = 0.f;
  for (int tq = tk; tq < kT; ++tq) s += __expf(p[(long)tq * kT] * scale - mx);
  float inv = 1.f / s;
  for (int tq = 0; tq < kT; ++tq) {
    float v = (tq >= tk) ? __expf(p[(long)tq * kT] * scale - mx) * inv : 0.f;
    po[(long)tq * kT] = f2bf(v);
  }
}

// ---------------------------------------------------------------- WMMA GEMM
// Block = 4 waves; each wave owns one 16-col N tile and 64 rows (4 M-tiles).
// The 64xK A panel is shared by all 4 waves: staged through a double-buffered
// LDS tile via async global->LDS copies; fragments re-read with ds_load_b128.
// A: MxK bf16 row-major. Bt: NxK bf16 row-major (pre-transposed B).
// batch z: A+=z*sA, Bt+=z*sB, out offset = (z/innerC)*sC0 + (z%innerC)*sC1.
// MODE 0: f32 out[m*ldc+n] = alpha*acc + bias + resid
// MODE 1: bf16 out[m*ldc+n]           (RELU optional, +bias)
// MODE 2: bf16 q/k layout out[(n>>6)*T*HS + m*HS + (n&63)]  (+bias)
// MODE 3: bf16 transposed out[n*ldc + m]                    (+bias)
template <int MODE, bool RELU>
__global__ void k_gemm(const __bf16* __restrict__ A, const __bf16* __restrict__ Bt,
                       const float* __restrict__ bias, void* __restrict__ Cout,
                       const float* __restrict__ resid,
                       int Ndim, int K, int lda, int ldb, int ldc,
                       long sA, long sB, long sC0, long sC1, int innerC, float alpha) {
  __shared__ __bf16 tileA[2][64 * kLdsStride];

  int lane = threadIdx.x;                 // 0..31
  int tid  = threadIdx.y * 32 + lane;     // 0..127
  int ntile = blockIdx.x * blockDim.y + threadIdx.y;
  int n0 = ntile * 16;                    // may be >= Ndim for tail tiles: keep
                                          // running (uniform barriers), mask stores
  int m0 = blockIdx.y * 64;
  int z  = blockIdx.z;
  A  += (long)z * sA;
  Bt += (long)z * sB;
  long coff = (long)(z / innerC) * sC0 + (long)(z % innerC) * sC1;

  floatx8 acc[4];
#pragma unroll
  for (int i = 0; i < 4; ++i)
#pragma unroll
    for (int r = 0; r < 8; ++r) acc[i][r] = 0.f;

  int arow   = lane & 15;
  int kshift = (lane >> 4) << 3;
  int brow   = n0 + (lane & 15);
  if (brow >= Ndim) brow = Ndim - 1;  // clamp loads for OOB tail tiles

  // Stage a 64-row x 32-col A chunk into LDS buffer `buf` (async).
  // 64 rows * 64B = 256 chunks of 16B; 128 threads -> 2 chunks each.
  auto stageA = [&](int k0, int buf) {
#pragma unroll
    for (int c = tid; c < 256; c += 128) {
      int row = c >> 2, q = c & 3;
      async_cp16(A + (long)(m0 + row) * lda + k0 + q * 8,
                 &tileA[buf][row * kLdsStride + q * 8]);
    }
  };

  stageA(0, 0);
  wait_async0();
  __syncthreads();

  int cur = 0;
  for (int k0 = 0; k0 < K; k0 += 32) {
    if (k0 + 32 < K) {
      stageA(k0 + 32, cur ^ 1);                                   // overlap copy
      __builtin_prefetch(Bt + (long)brow * ldb + k0 + 32, 0, 3);  // B next chunk
    }
    bf16x16 bf = load_frag(Bt + (long)brow * ldb + k0 + kshift);
#pragma unroll
    for (int i = 0; i < 4; ++i) {
      bf16x16 af = load_frag(&tileA[cur][(i * 16 + arow) * kLdsStride + kshift]);
      acc[i] = __builtin_amdgcn_wmma_f32_16x16x32_bf16(
          false, af, false, bf, (short)0, acc[i], false, false);
    }
    wait_async0();
    __syncthreads();
    cur ^= 1;
  }

  int ncol = n0 + (lane & 15);
  bool nok = ncol < Ndim;
  float bval = (bias != nullptr && nok) ? bias[ncol] : 0.f;
  int rbase = (lane >> 4) << 3;  // D layout: VGPR r -> row r (+8 for lanes 16-31)
#pragma unroll
  for (int i = 0; i < 4; ++i) {
#pragma unroll
    for (int r = 0; r < 8; ++r) {
      if (!nok) continue;
      int mm = m0 + i * 16 + rbase + r;
      float v = acc[i][r] * alpha + bval;
      if (MODE == 0) {
        float* C = (float*)Cout;
        long idx = coff + (long)mm * ldc + ncol;
        if (resid != nullptr) v += resid[idx];
        C[idx] = v;
      } else if (MODE == 1) {
        if (RELU) v = fmaxf(v, 0.f);
        ((__bf16*)Cout)[coff + (long)mm * ldc + ncol] = f2bf(v);
      } else if (MODE == 2) {
        ((__bf16*)Cout)[coff + (long)(ncol >> 6) * ((long)kT * kHS) +
                        (long)mm * kHS + (ncol & 63)] = f2bf(v);
      } else {
        ((__bf16*)Cout)[coff + (long)ncol * ldc + mm] = f2bf(v);
      }
    }
  }
}

// ---------------------------------------------------------------- host
extern "C" void kernel_launch(void* const* d_in, const int* in_sizes, int n_in,
                              void* d_out, int out_size, void* d_ws, size_t ws_size,
                              hipStream_t stream) {
  (void)in_sizes; (void)n_in; (void)out_size; (void)ws_size;
  const int*   x     = (const int*)d_in[0];
  const float* tok   = (const float*)d_in[1];
  const float* pos   = (const float*)d_in[2];
  const float* Wq    = (const float*)d_in[3];
  const float* bq    = (const float*)d_in[4];
  const float* Wk    = (const float*)d_in[5];
  const float* bk    = (const float*)d_in[6];
  const float* Wv    = (const float*)d_in[7];
  const float* bv    = (const float*)d_in[8];
  const float* Wo    = (const float*)d_in[9];
  const float* bo    = (const float*)d_in[10];
  const float* ln1g  = (const float*)d_in[11];
  const float* ln1b  = (const float*)d_in[12];
  const float* W1    = (const float*)d_in[13];
  const float* b1    = (const float*)d_in[14];
  const float* W2    = (const float*)d_in[15];
  const float* b2    = (const float*)d_in[16];
  const float* ln2g  = (const float*)d_in[17];
  const float* ln2b  = (const float*)d_in[18];
  const float* Wlm   = (const float*)d_in[19];
  const float* blm   = (const float*)d_in[20];
  float* out = (float*)d_out;

  char* ws = (char*)d_ws;
  size_t off = 0;
  auto alloc = [&](size_t bytes) {
    size_t r = off;
    off = (off + bytes + 255) & ~(size_t)255;
    return r;
  };
  float*  h      = (float*)(ws + alloc((size_t)kM * kC * 4));
  __bf16* a_bf   = (__bf16*)(ws + alloc((size_t)kM * kC * 2));
  __bf16* q_bf   = (__bf16*)(ws + alloc((size_t)kB * kH * kT * kHS * 2));
  __bf16* kk_bf  = (__bf16*)(ws + alloc((size_t)kB * kH * kT * kHS * 2));
  __bf16* vT_bf  = (__bf16*)(ws + alloc((size_t)kB * kH * kHS * kT * 2));
  __bf16* att_bf = (__bf16*)(ws + alloc((size_t)kM * kC * 2));
  __bf16* m1_bf  = (__bf16*)(ws + alloc((size_t)kM * kC * 2));
  __bf16* hb_bf  = (__bf16*)(ws + alloc((size_t)kM * kC * 2));
  float*  sc     = (float*)(ws + alloc((size_t)kB * kH * kT * kT * 4));
  __bf16* wsm_bf = (__bf16*)(ws + alloc((size_t)kB * kH * kT * kT * 2));
  __bf16* wtb    = (__bf16*)(ws + alloc((size_t)kV * kC * 2));  // reused for all weights

  dim3 blk(32, 4);  // 4 waves, one N-tile each

  // h = tok_emb[x] + pos_emb
  {
    long n = (long)kM * kC;
    k_embed<<<dim3((unsigned)((n + 255) / 256)), 256, 0, stream>>>(x, tok, pos, h);
  }

  for (int l = 0; l < kNL; ++l) {
    const long wqkv = (long)kH * kC * kHS;
    // ---- ln1
    k_layernorm<<<kM, 128, 0, stream>>>(h, ln1g + (long)l * kC, ln1b + (long)l * kC, a_bf);
    // ---- Q
    k_conv_qkv<<<dim3((unsigned)((wqkv + 255) / 256)), 256, 0, stream>>>(Wq + l * wqkv, wtb);
    k_gemm<2, false><<<dim3(kC / 64, kT / 64, kB), blk, 0, stream>>>(
        a_bf, wtb, bq + (long)l * kH * kHS, q_bf, nullptr,
        kC, kC, kC, kC, 0, (long)kT * kC, 0, (long)kH * kT * kHS, 0, 1, 1.f);
    // ---- K
    k_conv_qkv<<<dim3((unsigned)((wqkv + 255) / 256)), 256, 0, stream>>>(Wk + l * wqkv, wtb);
    k_gemm<2, false><<<dim3(kC / 64, kT / 64, kB), blk, 0, stream>>>(
        a_bf, wtb, bk + (long)l * kH * kHS, kk_bf, nullptr,
        kC, kC, kC, kC, 0, (long)kT * kC, 0, (long)kH * kT * kHS, 0, 1, 1.f);
    // ---- V (stored transposed: (B,H,HS,T))
    k_conv_qkv<<<dim3((unsigned)((wqkv + 255) / 256)), 256, 0, stream>>>(Wv + l * wqkv, wtb);
    k_gemm<3, false><<<dim3(kC / 64, kT / 64, kB), blk, 0, stream>>>(
        a_bf, wtb, bv + (long)l * kH * kHS, vT_bf, nullptr,
        kC, kC, kC, kC, kT, (long)kT * kC, 0, (long)kH * kHS * kT, 0, 1, 1.f);
    // ---- scores = q @ k^T  (batched over B*H)
    k_gemm<0, false><<<dim3(kT / 64, kT / 64, kB * kH), blk, 0, stream>>>(
        q_bf, kk_bf, nullptr, sc, nullptr,
        kT, kHS, kHS, kHS, kT, (long)kT * kHS, (long)kT * kHS, (long)kT * kT, 0, 1, 1.f);
    // ---- softmax over query axis with causal mask; scale = HS^-0.5
    k_softmax_q<<<dim3((kB * kH * kT + 255) / 256), 256, 0, stream>>>(sc, wsm_bf, 0.125f);
    // ---- att = w @ v  -> (B,T,C) bf16
    k_gemm<1, false><<<dim3(1, kT / 64, kB * kH), blk, 0, stream>>>(
        wsm_bf, vT_bf, nullptr, att_bf, nullptr,
        kHS, kT, kT, kT, kC, (long)kT * kT, (long)kHS * kT,
        (long)kT * kC, kHS, kH, 1.f);
    // ---- h += att @ Wo + bo
    k_transpose_w<<<dim3((kC * kC + 255) / 256), 256, 0, stream>>>(
        Wo + (long)l * kC * kC, wtb, kC, kC);
    k_gemm<0, false><<<dim3(kC / 64, kM / 64, 1), blk, 0, stream>>>(
        att_bf, wtb, bo + (long)l * kC, h, h,
        kC, kC, kC, kC, kC, 0, 0, 0, 0, 1, 1.f);
    // ---- ln2
    k_layernorm<<<kM, 128, 0, stream>>>(h, ln2g + (long)l * kC, ln2b + (long)l * kC, a_bf);
    // ---- m1 = relu(a @ W1 + b1)
    k_transpose_w<<<dim3((kC * kC + 255) / 256), 256, 0, stream>>>(
        W1 + (long)l * kC * kC, wtb, kC, kC);
    k_gemm<1, true><<<dim3(kC / 64, kM / 64, 1), blk, 0, stream>>>(
        a_bf, wtb, b1 + (long)l * kC, m1_bf, nullptr,
        kC, kC, kC, kC, kC, 0, 0, 0, 0, 1, 1.f);
    // ---- h += m1 @ W2 + b2
    k_transpose_w<<<dim3((kC * kC + 255) / 256), 256, 0, stream>>>(
        W2 + (long)l * kC * kC, wtb, kC, kC);
    k_gemm<0, false><<<dim3(kC / 64, kM / 64, 1), blk, 0, stream>>>(
        m1_bf, wtb, b2 + (long)l * kC, h, h,
        kC, kC, kC, kC, kC, 0, 0, 0, 0, 1, 1.f);
  }

  // ---- logits = h @ Wlm + blm
  {
    long n = (long)kM * kC;
    k_f32_to_bf16<<<dim3((unsigned)((n + 255) / 256)), 256, 0, stream>>>(h, hb_bf, n);
    long wn = (long)kC * kV;
    k_transpose_w<<<dim3((unsigned)((wn + 255) / 256)), 256, 0, stream>>>(Wlm, wtb, kC, kV);
    k_gemm<0, false><<<dim3((kV + 63) / 64, kM / 64, 1), blk, 0, stream>>>(
        hb_bf, wtb, blm, out, nullptr,
        kV, kC, kC, kC, kV, 0, 0, 0, 0, 1, 1.f);
  }
}
